// MultiHead_SelfAttention_2216203124953
// MI455X (gfx1250) — compile-verified, matrix operand order checked by reading, not run on hardware
//
#include <hip/hip_runtime.h>
#include <hip/hip_bf16.h>

// ---------------------------------------------------------------------------
// Multi-head self-attention on gfx1250 (MI455X).
//   B=4, S=2048, D=1024, H=16, hd=64
// All GEMMs via v_wmma_f32_16x16x32_bf16 (fp32 accumulate).
// Flash-attention K/V tiles (64x64 bf16) staged to LDS with the Tensor Data
// Mover (tensor_load_to_lds, TENSORcnt, double-buffered).
// ---------------------------------------------------------------------------

#define BATCH 4
#define SEQ   2048
#define DIM   1024
#define HEADS 16
#define HDIM  64
#define MROWS (BATCH * SEQ)   // 8192
#define KTILE 64              // keys per flash-attention iteration

typedef __attribute__((ext_vector_type(16))) __bf16 bf16x16;
typedef __attribute__((ext_vector_type(8)))  __bf16 bf16x8;
typedef __attribute__((ext_vector_type(8)))  float  f32x8;
typedef unsigned int __attribute__((ext_vector_type(4))) u32x4;
typedef int          __attribute__((ext_vector_type(8))) i32x8;
typedef int          __attribute__((ext_vector_type(4))) i32x4;

static __device__ __forceinline__ __bf16 f2bf(float f) { return (__bf16)f; }

// --------------------------- WMMA fragment loaders -------------------------
// 16-bit A-matrix 16x32 (MxK): lane L holds row M = L&15; element i maps to
// K = 8*(L>>4) + i + 8*(i>>3)  ->  two contiguous 8-element (16B) runs.
static __device__ __forceinline__ int frag_k(int lane, int i) {
    return ((lane >> 4) << 3) + i + ((i >> 3) << 3);
}

// A fragment from row-major bf16 source (also used for transposed-source B,
// where B[k][n] = src[n*ld + k]: identical addressing with n0 as row0).
static __device__ __forceinline__ bf16x16
load_a_bf16(const __bf16* __restrict__ src, int ld, int row0, int k0, int lane) {
    bf16x16 a;
    const size_t base = (size_t)(row0 + (lane & 15)) * ld + k0;
#pragma unroll
    for (int i = 0; i < 16; ++i) a[i] = src[base + frag_k(lane, i)];
    return a;
}

// B fragment (32x16, KxN) where B[k][n] = src[k*ld + n]  (natural source: P@V)
static __device__ __forceinline__ bf16x16
load_bn_bf16(const __bf16* __restrict__ src, int ld, int k0, int n0, int lane) {
    bf16x16 b;
    const int n = n0 + (lane & 15);
#pragma unroll
    for (int i = 0; i < 16; ++i)
        b[i] = src[(size_t)(k0 + frag_k(lane, i)) * ld + n];
    return b;
}

static __device__ __forceinline__ f32x8
wmma_bf16(bf16x16 a, bf16x16 b, f32x8 c) {
    return __builtin_amdgcn_wmma_f32_16x16x32_bf16(
        /*neg_a=*/false, a, /*neg_b=*/false, b,
        /*c_mod=*/(short)0, c, /*reuse_a=*/false, /*reuse_b=*/false);
}

// ---------------------------------------------------------------------------
// Tensor Data Mover: 2D bf16 tile load (tile_rows x rowlen) global -> LDS.
// D# layout per CDNA5 ISA (groups 2/3 zero = 2D).
// 6-arg builtin: (u32x4 g0, i32x8 g1, i32x4 g2, i32x4 g3, i32x8 pad, i32 cpol)
// ---------------------------------------------------------------------------
static __device__ __forceinline__ void
tdm_load_tile_bf16(const __bf16* gsrc, unsigned lds_off,
                   int tile_rows, int rowlen, int tensor_rows) {
    const unsigned long long ga = (unsigned long long)(uintptr_t)gsrc;
    u32x4 g0;
    g0[0] = 1u;                                   // count=1 (valid), user mode
    g0[1] = lds_off;                              // lds_addr (bytes)
    g0[2] = (unsigned)ga;                         // global_addr[31:0]
    g0[3] = (unsigned)(ga >> 32) | (2u << 30);    // global_addr[56:32] | type=2

    const unsigned long long stride1 = (unsigned long long)rowlen * tensor_rows;
    i32x8 g1;
    g1[0] = 0x00010000;                           // wg_mask=0, data_size=1 (2B)
    g1[1] = (rowlen & 0xFFFF) << 16;              // tensor_dim0[15:0]
    g1[2] = ((rowlen >> 16) & 0xFFFF) |           // tensor_dim0[31:16]
            ((tensor_rows & 0xFFFF) << 16);       // tensor_dim1[15:0]
    g1[3] = ((tensor_rows >> 16) & 0xFFFF) |      // tensor_dim1[31:16]
            ((rowlen & 0xFFFF) << 16);            // tile_dim0
    g1[4] = tile_rows & 0xFFFF;                   // tile_dim1 (tile_dim2=0)
    g1[5] = rowlen;                               // tensor_dim0_stride[31:0]
    g1[6] = (int)(((unsigned)stride1 & 0xFFFF) << 16); // str0[47:32]=0|str1[15:0]
    g1[7] = (int)(stride1 >> 16);                 // tensor_dim1_stride[47:16]

    i32x4 gz4 = {0, 0, 0, 0};
    i32x8 gz8 = {0, 0, 0, 0, 0, 0, 0, 0};
    __builtin_amdgcn_tensor_load_to_lds(g0, g1, gz4, gz4, gz8, 0);
}

// ---------------------------------------------------------------------------
// Kernel 0: bulk f32 -> bf16 conversion (8 elems/thread, 128-bit ld/st)
// ---------------------------------------------------------------------------
__global__ __launch_bounds__(256)
void cvt_f32_to_bf16_kernel(const float* __restrict__ src,
                            __bf16* __restrict__ dst, int n) {
    const int i = (blockIdx.x * blockDim.x + threadIdx.x) * 8;
    if (i + 8 > n) return;
    const float4 a = *(const float4*)(src + i);
    const float4 b = *(const float4*)(src + i + 4);
    bf16x8 o;
    o[0] = f2bf(a.x); o[1] = f2bf(a.y); o[2] = f2bf(a.z); o[3] = f2bf(a.w);
    o[4] = f2bf(b.x); o[5] = f2bf(b.y); o[6] = f2bf(b.z); o[7] = f2bf(b.w);
    *(bf16x8*)(dst + i) = o;
}

// ---------------------------------------------------------------------------
// Kernel 1: QKV projection.  y = x @ W^T + b   (M=8192, N=1024, K=1024)
// Output stored bf16 in [B, H, S, hd] layout.
// grid: (DIM/64, MROWS/128, 3)  block: 256 (8 waves, each 16 rows x 64 cols)
// ---------------------------------------------------------------------------
__global__ __launch_bounds__(256)
void qkv_proj_kernel(const __bf16* __restrict__ x,
                     const __bf16* __restrict__ Wq, const float* __restrict__ bq,
                     const __bf16* __restrict__ Wk, const float* __restrict__ bk,
                     const __bf16* __restrict__ Wv, const float* __restrict__ bv,
                     __bf16* __restrict__ Qo, __bf16* __restrict__ Ko,
                     __bf16* __restrict__ Vo) {
    const __bf16* W; const float* bias; __bf16* dst;
    if (blockIdx.z == 0)      { W = Wq; bias = bq; dst = Qo; }
    else if (blockIdx.z == 1) { W = Wk; bias = bk; dst = Ko; }
    else                      { W = Wv; bias = bv; dst = Vo; }

    const int lane = threadIdx.x & 31;
    const int wave = threadIdx.x >> 5;
    const int row0 = blockIdx.y * 128 + wave * 16;
    const int col0 = blockIdx.x * 64;

    f32x8 acc[4] = {};
#pragma unroll 2
    for (int k = 0; k < DIM; k += 32) {
        bf16x16 a = load_a_bf16(x, DIM, row0, k, lane);
#pragma unroll
        for (int t = 0; t < 4; ++t) {
            bf16x16 b = load_a_bf16(W, DIM, col0 + 16 * t, k, lane);
            acc[t] = wmma_bf16(a, b, acc[t]);
        }
    }

    // store in [B,H,S,hd] (bf16), add bias
#pragma unroll
    for (int t = 0; t < 4; ++t) {
        const int n = col0 + 16 * t + (lane & 15);
        const int h = n >> 6, d = n & 63;
        const float bv_ = bias[n];
#pragma unroll
        for (int r = 0; r < 8; ++r) {
            const int grow = row0 + r + ((lane >> 4) << 3); // b*S + s
            const int bb = grow >> 11, ss = grow & (SEQ - 1);
            dst[(((size_t)bb * HEADS + h) * SEQ + ss) * HDIM + d] =
                f2bf(acc[t][r] + bv_);
        }
    }
}

// ---------------------------------------------------------------------------
// Kernel 2: flash attention.  One wave = 16 queries of one (b,h).
// K/V tiles (64x64 bf16 = 8KB) staged to LDS by the TDM, double-buffered.
// grid: (SEQ/64, BATCH*HEADS)  block: 128 (4 waves)
// ---------------------------------------------------------------------------
__global__ __launch_bounds__(128)
void flash_attn_kernel(const __bf16* __restrict__ Q,
                       const __bf16* __restrict__ K,
                       const __bf16* __restrict__ V,
                       __bf16* __restrict__ ctx) {
    const int lane = threadIdx.x & 31;
    const int wave = threadIdx.x >> 5;
    const int bh   = blockIdx.y;
    const int q0   = blockIdx.x * 64 + wave * 16;

    const __bf16* Qb = Q + (size_t)bh * SEQ * HDIM;
    const __bf16* Kb = K + (size_t)bh * SEQ * HDIM;
    const __bf16* Vb = V + (size_t)bh * SEQ * HDIM;

    // double-buffered K/V tiles: [buf][K/V][KTILE*HDIM]  (32 KB)
    __shared__ __bf16 kvbuf[2][2][KTILE * HDIM];
    __shared__ __bf16 pbuf[4][16 * KTILE];   // per-wave P tile (16q x 64k, 8 KB)
    __bf16* P = pbuf[wave];

    // Q fragments for d=0..31 and d=32..63, held in registers for whole loop
    bf16x16 qa0 = load_a_bf16(Qb, HDIM, q0, 0,  lane);
    bf16x16 qa1 = load_a_bf16(Qb, HDIM, q0, 32, lane);

    f32x8 o[4] = {};
    float mrow[8], lrow[8];
#pragma unroll
    for (int r = 0; r < 8; ++r) { mrow[r] = -1e30f; lrow[r] = 0.f; }

    const float scale = 0.125f; // 1/sqrt(64)

    // preload tile 0 (TDM issues once per wave; restrict to wave 0)
    if (wave == 0) {
        tdm_load_tile_bf16(Kb, (unsigned)(uintptr_t)&kvbuf[0][0][0], KTILE, HDIM, SEQ);
        tdm_load_tile_bf16(Vb, (unsigned)(uintptr_t)&kvbuf[0][1][0], KTILE, HDIM, SEQ);
    }

    for (int kt = 0; kt < SEQ; kt += KTILE) {
        const int buf = (kt >> 6) & 1;

        if (wave == 0) {
            if (kt + KTILE < SEQ) {
                // issue next tile into the other buffer, then wait for current
                tdm_load_tile_bf16(Kb + (size_t)(kt + KTILE) * HDIM,
                                   (unsigned)(uintptr_t)&kvbuf[buf ^ 1][0][0],
                                   KTILE, HDIM, SEQ);
                tdm_load_tile_bf16(Vb + (size_t)(kt + KTILE) * HDIM,
                                   (unsigned)(uintptr_t)&kvbuf[buf ^ 1][1][0],
                                   KTILE, HDIM, SEQ);
                __builtin_amdgcn_s_wait_tensorcnt(2);  // current 2 ops retired
            } else {
                __builtin_amdgcn_s_wait_tensorcnt(0);
            }
        }
        __syncthreads();  // K/V tile visible to all waves

        const __bf16* Kt = &kvbuf[buf][0][0];
        const __bf16* Vt = &kvbuf[buf][1][0];

        // ---- S = Q K^T for 16q x 64keys (four 16-key column tiles) ----
        f32x8 s[4];
#pragma unroll
        for (int f = 0; f < 4; ++f) {
            bf16x16 b0 = load_a_bf16(Kt, HDIM, 16 * f, 0,  lane);
            bf16x16 b1 = load_a_bf16(Kt, HDIM, 16 * f, 32, lane);
            f32x8 acc = {};
            acc = wmma_bf16(qa0, b0, acc);
            acc = wmma_bf16(qa1, b1, acc);
            s[f] = acc;
        }

        // ---- online softmax (per row; rows live in 16-lane groups) ----
#pragma unroll
        for (int r = 0; r < 8; ++r) {
            float v0 = s[0][r] * scale;
            float v1 = s[1][r] * scale;
            float v2 = s[2][r] * scale;
            float v3 = s[3][r] * scale;
            float mx = fmaxf(fmaxf(v0, v1), fmaxf(v2, v3));
#pragma unroll
            for (int off = 8; off >= 1; off >>= 1)
                mx = fmaxf(mx, __shfl_xor(mx, off, 16));
            const float mnew = fmaxf(mrow[r], mx);
            const float alpha = __expf(mrow[r] - mnew);
            mrow[r] = mnew;
            v0 = __expf(v0 - mnew);
            v1 = __expf(v1 - mnew);
            v2 = __expf(v2 - mnew);
            v3 = __expf(v3 - mnew);
            float rs = (v0 + v1) + (v2 + v3);
#pragma unroll
            for (int off = 8; off >= 1; off >>= 1)
                rs += __shfl_xor(rs, off, 16);
            lrow[r] = lrow[r] * alpha + rs;
            s[0][r] = v0; s[1][r] = v1; s[2][r] = v2; s[3][r] = v3;
#pragma unroll
            for (int t = 0; t < 4; ++t) o[t][r] *= alpha;
        }

        // ---- stage P through LDS: C-layout -> row-major (A-layout) ----
#pragma unroll
        for (int r = 0; r < 8; ++r) {
            const int m = r + ((lane >> 4) << 3);
#pragma unroll
            for (int f = 0; f < 4; ++f)
                P[m * KTILE + 16 * f + (lane & 15)] = f2bf(s[f][r]);
        }
        __syncthreads();  // all 4 waves run identical trip counts

        bf16x16 pa0 = load_a_bf16(P, KTILE, 0, 0,  lane);
        bf16x16 pa1 = load_a_bf16(P, KTILE, 0, 32, lane);

        // ---- O += P V  (16q x 64d, four 16-col tiles; K-dim 64) ----
#pragma unroll
        for (int t = 0; t < 4; ++t) {
            bf16x16 b0 = load_bn_bf16(Vt, HDIM, 0,  16 * t, lane);
            bf16x16 b1 = load_bn_bf16(Vt, HDIM, 32, 16 * t, lane);
            o[t] = wmma_bf16(pa0, b0, o[t]);
            o[t] = wmma_bf16(pa1, b1, o[t]);
        }
        __syncthreads();  // reads done before next TDM overwrite / P rewrite
    }

    // ---- normalize and write ctx (bf16, [B,S,D]) ----
    const int bb = bh / HEADS, hh = bh % HEADS;
#pragma unroll
    for (int t = 0; t < 4; ++t) {
        const int d = hh * HDIM + 16 * t + (lane & 15);
#pragma unroll
        for (int r = 0; r < 8; ++r) {
            const int s_ = q0 + r + ((lane >> 4) << 3);
            ctx[((size_t)bb * SEQ + s_) * DIM + d] = f2bf(o[t][r] / lrow[r]);
        }
    }
}

// ---------------------------------------------------------------------------
// Kernel 3: output projection.  out = ctx @ Wo^T + bo   (fp32 out)
// grid: (DIM/64, MROWS/128)  block: 256
// ---------------------------------------------------------------------------
__global__ __launch_bounds__(256)
void out_proj_kernel(const __bf16* __restrict__ ctx,
                     const __bf16* __restrict__ Wo,
                     const float* __restrict__ bo,
                     float* __restrict__ out) {
    const int lane = threadIdx.x & 31;
    const int wave = threadIdx.x >> 5;
    const int row0 = blockIdx.y * 128 + wave * 16;
    const int col0 = blockIdx.x * 64;

    f32x8 acc[4] = {};
#pragma unroll 2
    for (int k = 0; k < DIM; k += 32) {
        bf16x16 a = load_a_bf16(ctx, DIM, row0, k, lane);
#pragma unroll
        for (int t = 0; t < 4; ++t) {
            bf16x16 b = load_a_bf16(Wo, DIM, col0 + 16 * t, k, lane);
            acc[t] = wmma_bf16(a, b, acc[t]);
        }
    }

#pragma unroll
    for (int t = 0; t < 4; ++t) {
        const int n = col0 + 16 * t + (lane & 15);
        const float bv_ = bo[n];
#pragma unroll
        for (int r = 0; r < 8; ++r) {
            const int m = row0 + r + ((lane >> 4) << 3);
            out[(size_t)m * DIM + n] = acc[t][r] + bv_;
        }
    }
}

// ---------------------------------------------------------------------------
extern "C" void kernel_launch(void* const* d_in, const int* in_sizes, int n_in,
                              void* d_out, int out_size, void* d_ws, size_t ws_size,
                              hipStream_t stream) {
    const float* x  = (const float*)d_in[0];
    const float* Wq = (const float*)d_in[1];
    const float* bq = (const float*)d_in[2];
    const float* Wk = (const float*)d_in[3];
    const float* bk = (const float*)d_in[4];
    const float* Wv = (const float*)d_in[5];
    const float* bv = (const float*)d_in[6];
    const float* Wo = (const float*)d_in[7];
    const float* bo = (const float*)d_in[8];
    float* out = (float*)d_out;

    // workspace layout (bf16 elements):
    //   xbf (8M) | Wq,Wk,Wv,Wo (1M each) | Q,K,V (8M each) | ctx (8M)
    const size_t xe = (size_t)MROWS * DIM;   // 8M
    const size_t we = (size_t)DIM * DIM;     // 1M
    __bf16* xbf = (__bf16*)d_ws;
    __bf16* Wqb = xbf + xe;
    __bf16* Wkb = Wqb + we;
    __bf16* Wvb = Wkb + we;
    __bf16* Wob = Wvb + we;
    __bf16* Qw  = Wob + we;
    __bf16* Kw  = Qw + xe;
    __bf16* Vw  = Kw + xe;
    __bf16* ctx = Vw + xe;

    // 0) convert x and weights to bf16 (8 elems per thread)
    cvt_f32_to_bf16_kernel<<<(int)(xe / (256 * 8)), 256, 0, stream>>>(x, xbf, (int)xe);
    cvt_f32_to_bf16_kernel<<<(int)(we / (256 * 8)), 256, 0, stream>>>(Wq, Wqb, (int)we);
    cvt_f32_to_bf16_kernel<<<(int)(we / (256 * 8)), 256, 0, stream>>>(Wk, Wkb, (int)we);
    cvt_f32_to_bf16_kernel<<<(int)(we / (256 * 8)), 256, 0, stream>>>(Wv, Wvb, (int)we);
    cvt_f32_to_bf16_kernel<<<(int)(we / (256 * 8)), 256, 0, stream>>>(Wo, Wob, (int)we);

    // 1) QKV projection
    dim3 gQKV(DIM / 64, MROWS / 128, 3);
    qkv_proj_kernel<<<gQKV, 256, 0, stream>>>(xbf, Wqb, bq, Wkb, bk, Wvb, bv,
                                              Qw, Kw, Vw);

    // 2) flash attention (TDM-staged K/V)
    dim3 gAtt(SEQ / 64, BATCH * HEADS);
    flash_attn_kernel<<<gAtt, 128, 0, stream>>>(Qw, Kw, Vw, ctx);

    // 3) output projection
    dim3 gOut(DIM / 64, MROWS / 128);
    out_proj_kernel<<<gOut, 256, 0, stream>>>(ctx, Wob, bo, out);
}
